// MultiHeadAttention_89919435309734
// MI455X (gfx1250) — matrix operand order repaired and checked by reading, not compiled, not run on hardware
//
#include <hip/hip_runtime.h>
#include <math.h>

// ---------------------------------------------------------------------------
// MI455X (gfx1250) multi-head attention, bf16 WMMA path.
// wave32; v_wmma_f32_16x16x32_bf16 for all matmuls; flash-attention with the
// "transposed" formulation so softmax stats are lane-uniform.
// ---------------------------------------------------------------------------

#define B_  4
#define S_  2048
#define E_  1024
#define H_  16
#define DK_ 64
#define BS_ 8192  // B_*S_

typedef __bf16 bf16_t;
typedef __attribute__((ext_vector_type(16))) __bf16 v16bf;
typedef __attribute__((ext_vector_type(8)))  __bf16 v8bf;
typedef __attribute__((ext_vector_type(8)))  float  v8f;

// workspace layout (bf16 elements)
constexpr size_t N_XB  = (size_t)BS_ * E_;   // 8,388,608
constexpr size_t N_W   = (size_t)E_ * E_;    // 1,048,576
constexpr size_t OFF_XB = 0;
constexpr size_t OFF_WQ = OFF_XB + N_XB;
constexpr size_t OFF_WK = OFF_WQ + N_W;
constexpr size_t OFF_WV = OFF_WK + N_W;
constexpr size_t OFF_WO = OFF_WV + N_W;
constexpr size_t OFF_Q  = OFF_WO + N_W;
constexpr size_t OFF_K  = OFF_Q  + N_XB;
constexpr size_t OFF_V  = OFF_K  + N_XB;   // V stored transposed [B,H,DK,S]
constexpr size_t OFF_AO = OFF_V  + N_XB;   // attention output [B,S,E] bf16

static __device__ inline unsigned int pack2_bf16(float a, float b) {
  union { __bf16 h[2]; unsigned int u; } x;
  x.h[0] = (__bf16)a; x.h[1] = (__bf16)b;
  return x.u;
}

static __device__ inline v8f wmma_bf16(v16bf a, v16bf b, v8f c) {
  // D = A(16x32 bf16) * B(32x16 bf16) + C(16x16 f32)
  return __builtin_amdgcn_wmma_f32_16x16x32_bf16(
      /*neg_a=*/false, a, /*neg_b=*/false, b,
      /*c_mod=*/(short)0, c, /*reuse_a=*/false, /*reuse_b=*/false);
}

// A-fragment (16x32 bf16, rows M contiguous-K in memory), base = &A[row0][k0]
static __device__ inline v16bf load_a_frag(const bf16_t* base, int lda,
                                           int half, int ln) {
  const bf16_t* p = base + (size_t)ln * lda + half * 16;
  v8bf lo = *(const v8bf*)(p);        // K = 16*half + 0..7
  v8bf hi = *(const v8bf*)(p + 8);    // K = 16*half + 8..15
  return __builtin_shufflevector(lo, hi, 0,1,2,3,4,5,6,7,8,9,10,11,12,13,14,15);
}

// B-fragment (32x16 bf16; column n = K-contiguous row n of source), base=&W[n0][k0]
static __device__ inline v16bf load_b_frag(const bf16_t* base, int ldb,
                                           int half, int ln) {
  const bf16_t* p = base + (size_t)ln * ldb + half * 16;
  return *(const v16bf*)(p);          // lane<16: K=0..15, lane>=16: K=16..31
}

// ---------------------------------------------------------------------------
// fp32 -> bf16 conversion (grid-stride, 2 elements per thread-iter)
// ---------------------------------------------------------------------------
__global__ void cvt_f32_bf16(const float* __restrict__ s,
                             bf16_t* __restrict__ d, int n2) {
  int i = blockIdx.x * blockDim.x + threadIdx.x;
  int stride = gridDim.x * blockDim.x;
  for (; i < n2; i += stride) {
    float2 f = ((const float2*)s)[i];
    ((unsigned int*)d)[i] = pack2_bf16(f.x, f.y);
  }
}

// ---------------------------------------------------------------------------
// Generic  C[M x N] = A[M x K] * W[N x K]^T  (bf16 in, f32 accum).
// Block: 256 thr = 8 waves arranged 4(M) x 2(N); wave tile 32x32.
// EPI 0: store bf16 to [B,H,S,DK]      (Q, K)
// EPI 1: store bf16 to [B,H,DK,S]      (V transposed)
// EPI 2: store f32  to [M, E]          (final output)
// ---------------------------------------------------------------------------
template <int EPI>
__global__ __launch_bounds__(256) void gemm_nt_bf16(const bf16_t* __restrict__ A,
                                                    const bf16_t* __restrict__ W,
                                                    void* __restrict__ outp) {
  const int lane = threadIdx.x & 31;
  const int half = lane >> 4;
  const int ln   = lane & 15;
  const int wave = threadIdx.x >> 5;
  const int m0 = blockIdx.x * 128 + (wave >> 1) * 32;
  const int n0 = blockIdx.y * 64  + (wave & 1) * 32;

  v8f zero = {0.f,0.f,0.f,0.f,0.f,0.f,0.f,0.f};
  v8f c[2][2] = {{zero, zero}, {zero, zero}};

  for (int k0 = 0; k0 < E_; k0 += 32) {
    v16bf a[2], b[2];
    #pragma unroll
    for (int i = 0; i < 2; ++i)
      a[i] = load_a_frag(A + (size_t)(m0 + 16*i) * E_ + k0, E_, half, ln);
    #pragma unroll
    for (int j = 0; j < 2; ++j)
      b[j] = load_b_frag(W + (size_t)(n0 + 16*j) * E_ + k0, E_, half, ln);
    #pragma unroll
    for (int i = 0; i < 2; ++i)
      #pragma unroll
      for (int j = 0; j < 2; ++j)
        c[i][j] = wmma_bf16(a[i], b[j], c[i][j]);
  }

  #pragma unroll
  for (int i = 0; i < 2; ++i) {
    #pragma unroll
    for (int j = 0; j < 2; ++j) {
      const int col = n0 + 16*j + ln;
      if constexpr (EPI == 2) {
        float* out = (float*)outp;
        #pragma unroll
        for (int r = 0; r < 8; ++r) {
          int row = m0 + 16*i + 8*half + r;
          out[(size_t)row * E_ + col] = c[i][j][r];
        }
      } else if constexpr (EPI == 0) {
        bf16_t* out = (bf16_t*)outp;
        const int h = col >> 6, dk = col & 63;
        #pragma unroll
        for (int r = 0; r < 8; ++r) {
          int row = m0 + 16*i + 8*half + r;      // row = b*S + s
          int bb = row >> 11, s = row & (S_ - 1);
          out[(((size_t)(bb * H_ + h)) * S_ + s) * DK_ + dk] = (bf16_t)c[i][j][r];
        }
      } else { // EPI == 1 : V transposed, pack pairs along s
        bf16_t* out = (bf16_t*)outp;
        const int h = col >> 6, dk = col & 63;
        #pragma unroll
        for (int r = 0; r < 8; r += 2) {
          int row = m0 + 16*i + 8*half + r;
          int bb = row >> 11, s = row & (S_ - 1);
          size_t idx = (((size_t)(bb * H_ + h)) * DK_ + dk) * S_ + s;
          *(unsigned int*)(out + idx) = pack2_bf16(c[i][j][r], c[i][j][r+1]);
        }
      }
    }
  }
}

// ---------------------------------------------------------------------------
// Flash attention (transposed formulation):
//   S^T = K x Q^T      (A = K rows,  B = Q^T; lane owns one query column)
//   O^T = V^T x P^T    (A = V^T rows, B = P^T built via xor-16 shuffles)
// 1 wave = 16 queries; block = 8 waves = 128 queries; kv processed 64 at a time.
// ---------------------------------------------------------------------------
__global__ __launch_bounds__(256) void flash_attn_kernel(const bf16_t* __restrict__ Q,
                                                         const bf16_t* __restrict__ Kb,
                                                         const bf16_t* __restrict__ Vt,
                                                         bf16_t* __restrict__ AO) {
  const int lane = threadIdx.x & 31;
  const int half = lane >> 4;
  const int ln   = lane & 15;
  const int wave = threadIdx.x >> 5;
  const int bh = blockIdx.y;
  const int q0 = blockIdx.x * 128 + wave * 16;

  const bf16_t* Qh = Q  + (size_t)bh * S_ * DK_;
  const bf16_t* Kh = Kb + (size_t)bh * S_ * DK_;
  const bf16_t* Vh = Vt + (size_t)bh * DK_ * S_;

  // Q^T B-fragments (contraction dim = dk, N dim = query); loaded once.
  v16bf qf[2];
  #pragma unroll
  for (int ks = 0; ks < 2; ++ks)
    qf[ks] = *(const v16bf*)(Qh + (size_t)(q0 + ln) * DK_ + ks * 32 + half * 16);

  v8f zero = {0.f,0.f,0.f,0.f,0.f,0.f,0.f,0.f};
  v8f ot[4] = {zero, zero, zero, zero};     // O^T accum: tile d -> dk rows 16d..16d+15
  float m_run = -1e30f, l_run = 0.f;        // lane-scalar softmax stats (per query)

  for (int t0 = 0; t0 < S_; t0 += 64) {
    // ---- scores^T : 4 kv-tiles of 16, each 2 WMMAs over dk
    v8f st[4];
    #pragma unroll
    for (int i = 0; i < 4; ++i) {
      v8f acc = zero;
      #pragma unroll
      for (int ks = 0; ks < 2; ++ks) {
        v16bf kf = load_a_frag(Kh + (size_t)(t0 + 16*i) * DK_ + ks * 32,
                               DK_, half, ln);
        acc = wmma_bf16(kf, qf[ks], acc);
      }
      st[i] = acc;
    }

    // ---- online softmax (per-query stats are lane-scalar, xor-16 partner combine)
    float bm = -1e30f;
    #pragma unroll
    for (int i = 0; i < 4; ++i)
      #pragma unroll
      for (int r = 0; r < 8; ++r) {
        st[i][r] *= 0.125f;                 // 1/sqrt(Dk)
        bm = fmaxf(bm, st[i][r]);
      }
    bm = fmaxf(bm, __shfl_xor(bm, 16, 32));
    float m_new = fmaxf(m_run, bm);
    float alpha = __expf(m_run - m_new);
    float ps = 0.f;
    #pragma unroll
    for (int i = 0; i < 4; ++i)
      #pragma unroll
      for (int r = 0; r < 8; ++r) {
        float pv = __expf(st[i][r] - m_new);
        st[i][r] = pv;
        ps += pv;
      }
    ps += __shfl_xor(ps, 16, 32);
    l_run = l_run * alpha + ps;
    m_run = m_new;
    #pragma unroll
    for (int d = 0; d < 4; ++d)
      #pragma unroll
      for (int r = 0; r < 8; ++r)
        ot[d][r] *= alpha;

    // ---- P^T B-fragments from D-layout via xor-16 shuffles, then O^T += V^T * P^T
    #pragma unroll
    for (int j = 0; j < 2; ++j) {           // kv k-step: rows t0+32j .. +31
      float vl[8], vh[8];
      #pragma unroll
      for (int r = 0; r < 8; ++r) {
        float oa = __shfl_xor(st[2*j][r],   16, 32);
        float ob = __shfl_xor(st[2*j+1][r], 16, 32);
        // lane<16 wants tile 2j rows 0..15 (own rows 0..7 + partner rows 8..15)
        // lane>=16 wants tile 2j+1 rows 0..15 (partner rows 0..7 + own rows 8..15)
        vl[r] = half ? ob : st[2*j][r];
        vh[r] = half ? st[2*j+1][r] : oa;
      }
      union { unsigned int u[8]; v16bf v; } pt;
      #pragma unroll
      for (int r = 0; r < 8; r += 2) {
        pt.u[r/2]     = pack2_bf16(vl[r], vl[r+1]);   // K pairs 0..7
        pt.u[4 + r/2] = pack2_bf16(vh[r], vh[r+1]);   // K pairs 8..15
      }
      #pragma unroll
      for (int d = 0; d < 4; ++d) {
        v16bf vf = load_a_frag(Vh + (size_t)(16*d) * S_ + t0 + 32*j,
                               S_, half, ln);
        ot[d] = wmma_bf16(vf, pt.v, ot[d]);
      }
    }
  }

  // ---- normalize and store to AO [B,S,E] as bf16, packed pairs along dk
  const float inv_l = 1.f / l_run;
  const int b = bh >> 4, h = bh & 15;
  const int s = q0 + ln;
  bf16_t* dst = AO + ((size_t)(b * S_ + s)) * E_ + h * DK_;
  #pragma unroll
  for (int d = 0; d < 4; ++d)
    #pragma unroll
    for (int r = 0; r < 8; r += 2) {
      int dk = 16*d + 8*half + r;
      *(unsigned int*)(dst + dk) =
          pack2_bf16(ot[d][r] * inv_l, ot[d][r+1] * inv_l);
    }
}

// ---------------------------------------------------------------------------
extern "C" void kernel_launch(void* const* d_in, const int* in_sizes, int n_in,
                              void* d_out, int out_size, void* d_ws, size_t ws_size,
                              hipStream_t stream) {
  (void)in_sizes; (void)n_in; (void)out_size; (void)ws_size;
  const float* x  = (const float*)d_in[0];
  const float* Wq = (const float*)d_in[1];
  const float* Wk = (const float*)d_in[2];
  const float* Wv = (const float*)d_in[3];
  const float* Wo = (const float*)d_in[4];

  bf16_t* ws  = (bf16_t*)d_ws;
  bf16_t* xb  = ws + OFF_XB;
  bf16_t* wqb = ws + OFF_WQ;
  bf16_t* wkb = ws + OFF_WK;
  bf16_t* wvb = ws + OFF_WV;
  bf16_t* wob = ws + OFF_WO;
  bf16_t* Qb  = ws + OFF_Q;
  bf16_t* Kb  = ws + OFF_K;
  bf16_t* Vb  = ws + OFF_V;
  bf16_t* AOb = ws + OFF_AO;

  // fp32 -> bf16
  cvt_f32_bf16<<<2048, 256, 0, stream>>>(x,  xb,  (int)(N_XB / 2));
  cvt_f32_bf16<<<512,  256, 0, stream>>>(Wq, wqb, (int)(N_W / 2));
  cvt_f32_bf16<<<512,  256, 0, stream>>>(Wk, wkb, (int)(N_W / 2));
  cvt_f32_bf16<<<512,  256, 0, stream>>>(Wv, wvb, (int)(N_W / 2));
  cvt_f32_bf16<<<512,  256, 0, stream>>>(Wo, wob, (int)(N_W / 2));

  // QKV projections (x @ W^T), V stored transposed
  dim3 gg(BS_ / 128, E_ / 64);
  gemm_nt_bf16<0><<<gg, 256, 0, stream>>>(xb, wqb, Qb);
  gemm_nt_bf16<0><<<gg, 256, 0, stream>>>(xb, wkb, Kb);
  gemm_nt_bf16<1><<<gg, 256, 0, stream>>>(xb, wvb, Vb);

  // flash attention: grid (query tiles of 128, B*H)
  flash_attn_kernel<<<dim3(S_ / 128, B_ * H_), 256, 0, stream>>>(Qb, Kb, Vb, AOb);

  // output projection -> fp32 d_out
  gemm_nt_bf16<2><<<gg, 256, 0, stream>>>(AOb, wob, d_out);
}
